// Block_51015621542219
// MI455X (gfx1250) — matrix operand order repaired and checked by reading, not compiled, not run on hardware
//
#include <hip/hip_runtime.h>

// ---------------- problem constants (match reference) ----------------
constexpr int B    = 512;
constexpr int NPG  = 64;
constexpr int CNUM = 8;
constexpr int MPG  = 128;
constexpr int CN   = 256;
constexpr int CE   = 64;
constexpr int HN   = 4;
constexpr int HE   = 4;
constexpr int EPG  = MPG + CNUM + NPG;     // 200
constexpr int NN   = B * NPG;              // 32768 nodes
constexpr int NBR  = B * MPG;              // 65536 br rows
constexpr int NEI  = B * EPG;              // 102400 edge-inform rows
constexpr int NBE  = B * MPG * 4;          // 262144 br edges

// ---------------- vector types (POD ext-vectors, union-safe) ----------------
typedef __attribute__((ext_vector_type(16))) __bf16        v16bf;
typedef __attribute__((ext_vector_type(8)))  float         v8f;
typedef __attribute__((ext_vector_type(4)))  float         v4f;
typedef __attribute__((ext_vector_type(4)))  unsigned int  v4u;
typedef __attribute__((ext_vector_type(8)))  unsigned int  v8u;
union BF16x16 { unsigned short u[16]; v4u q[2]; v16bf v; };

// ---------------- helpers ----------------
__device__ __forceinline__ unsigned short f2bf(float f) {
  unsigned int u = __float_as_uint(f);
  unsigned int r = u + 0x7FFFu + ((u >> 16) & 1u);   // RNE truncation
  return (unsigned short)(r >> 16);
}
__device__ __forceinline__ unsigned int pack2(float a, float b) {
  return (unsigned int)f2bf(a) | ((unsigned int)f2bf(b) << 16);
}
// order-preserving float<->uint for LDS atomicMax (ds_max_u32)
__device__ __forceinline__ unsigned int f2ord(float f) {
  unsigned int u = __float_as_uint(f);
  return (u & 0x80000000u) ? ~u : (u | 0x80000000u);
}
__device__ __forceinline__ float ord2f(unsigned int u) {
  unsigned int r = (u & 0x80000000u) ? (u & 0x7FFFFFFFu) : ~u;
  return __uint_as_float(r);
}

// ---------------- TDM: 1-D tile load Global->LDS (D# per ISA 8.3/8.4) -------
// elems = #fp32 elements (must fit tile_dim0: <= 65535). Issued per-wave.
__device__ __forceinline__ void tdm_load_1d(const float* gsrc, float* lds_dst,
                                            unsigned elems) {
  unsigned long long ga = (unsigned long long)gsrc;
  unsigned lds = (unsigned)(unsigned long long)lds_dst;  // flat LDS addr low 32 = LDS offset
  v4u g0;
  g0[0] = 1u;                                            // count=1, user mode
  g0[1] = lds;                                           // lds_addr
  g0[2] = (unsigned)(ga & 0xFFFFFFFFull);                // global_addr[31:0]
  g0[3] = (unsigned)((ga >> 32) & 0x01FFFFFFull)         // global_addr[56:32]
          | (2u << 30);                                  // type=2 (image)
  v8u g1;
  g1[0] = 0x00020000u;                                   // data_size=2 (4B), mask=0
  g1[1] = (elems & 0xFFFFu) << 16;                       // tensor_dim0[15:0]
  g1[2] = (elems >> 16) | (1u << 16);                    // tensor_dim0[31:16] | tensor_dim1=1
  g1[3] = (elems & 0xFFFFu) << 16;                       // tensor_dim1 hi=0 | tile_dim0=elems
  g1[4] = 1u;                                            // tile_dim1=1, tile_dim2=0
  g1[5] = elems;                                         // tensor_dim0_stride[31:0]
  g1[6] = 0u;                                            // stride0 hi | stride1 lo
  g1[7] = 0u;                                            // stride1 hi
  asm volatile("tensor_load_to_lds %0, %1" :: "s"(g0), "s"(g1) : "memory");
}

// ---------------- WMMA GEMM: Y[M,N] = X[M,K] @ W[K,N] (fp32 io, bf16 mul) ----
// 256 thr = 8 waves; block tile 64x64; wave -> (wm, wn0..wn0+1): 2 WMMA tiles,
// shared A fragment. K,N compile-time; M % 64 == 0 (true for all calls).
template <int K, int N>
__global__ void __launch_bounds__(256)
wmma_gemm_kernel(const float* __restrict__ X, const float* __restrict__ W,
                 float* __restrict__ Y, int M) {
  constexpr int LDK = 40;                      // padded row stride (ushorts)
  __shared__ unsigned short As[64 * LDK];      // [m][k] bf16 bits
  __shared__ unsigned short BsT[64 * LDK];     // [n][k] bf16 bits (transposed)

  const int tid  = threadIdx.x;
  const int lane = tid & 31;
  const int wave = tid >> 5;
  const int wm   = wave >> 1;                  // 0..3
  const int wn0  = (wave & 1) * 2;             // 0 or 2
  const int m0 = blockIdx.y * 64;
  const int n0 = blockIdx.x * 64;
  const int half = lane >> 4;
  const int l15  = lane & 15;

  const int sr = tid >> 2;                     // staging row / n (0..63)
  const int sc = (tid & 3) * 8;                // staging k chunk (0,8,16,24)

  v8f acc0 = {}, acc1 = {};
  for (int kk = 0; kk < K; kk += 32) {
    // ---- stage A (64x32) row-major, coalesced float4 loads ----
    {
      const v4f* src = (const v4f*)&X[(size_t)(m0 + sr) * K + kk + sc];
      v4f f0 = src[0], f1 = src[1];
      v4u p;
      p[0] = pack2(f0[0], f0[1]); p[1] = pack2(f0[2], f0[3]);
      p[2] = pack2(f1[0], f1[1]); p[3] = pack2(f1[2], f1[3]);
      *(v4u*)&As[sr * LDK + sc] = p;
    }
    // ---- stage B transposed: BsT[n][k] <- W[kk+k][n0+n] ----
    {
      unsigned short t[8];
#pragma unroll
      for (int j = 0; j < 8; ++j)
        t[j] = f2bf(W[(size_t)(kk + sc + j) * N + (n0 + sr)]);
      v4u p;
      p[0] = (unsigned)t[0] | ((unsigned)t[1] << 16);
      p[1] = (unsigned)t[2] | ((unsigned)t[3] << 16);
      p[2] = (unsigned)t[4] | ((unsigned)t[5] << 16);
      p[3] = (unsigned)t[6] | ((unsigned)t[7] << 16);
      *(v4u*)&BsT[sr * LDK + sc] = p;
    }
    if (kk + 32 < K) {   // hint next tiles (global_prefetch_b8)
      __builtin_prefetch(&X[(size_t)(m0 + sr) * K + kk + 32 + sc], 0, 1);
      __builtin_prefetch(&W[(size_t)(kk + 32 + sc) * N + n0 + sr], 0, 1);
    }
    __syncthreads();

    // ---- fragments: contiguous 128b LDS loads in contiguous k order (A and B match) ----
    // A: a.u[0..15] -> K=16*half..+15
    BF16x16 a, b0, b1;
    const unsigned short* arow = &As[(wm * 16 + l15) * LDK];
    a.q[0] = *(const v4u*)&arow[16 * half];
    a.q[1] = *(const v4u*)&arow[16 * half + 8];
    // B: b.u[0..15] -> K=16*half..+15 of column n (BsT row n)
    const unsigned short* b0row = &BsT[(wn0 * 16 + l15) * LDK + 16 * half];
    const unsigned short* b1row = &BsT[((wn0 + 1) * 16 + l15) * LDK + 16 * half];
    b0.q[0] = ((const v4u*)b0row)[0]; b0.q[1] = ((const v4u*)b0row)[1];
    b1.q[0] = ((const v4u*)b1row)[0]; b1.q[1] = ((const v4u*)b1row)[1];

    acc0 = __builtin_amdgcn_wmma_f32_16x16x32_bf16(false, a.v, false, b0.v,
                                                   (short)0, acc0, false, false);
    acc1 = __builtin_amdgcn_wmma_f32_16x16x32_bf16(false, a.v, false, b1.v,
                                                   (short)0, acc1, false, false);
    __syncthreads();
  }
  // D layout: VGPR r -> row r + 8*half
  const int col = n0 + wn0 * 16 + l15;
#pragma unroll
  for (int r = 0; r < 8; ++r) {
    int row = m0 + wm * 16 + r + 8 * half;
    Y[(size_t)row * N + col]      = acc0[r];
    Y[(size_t)row * N + col + 16] = acc1[r];
  }
}

// ---------------- per-graph multi-head attention (block = 1 graph) ----------
// STAGE: TDM-load this graph's contiguous q/k/v blocks into LDS first.
template <int NODES, int EDGES, int HEADS, int OH, bool HAS_EF, bool STAGE>
__global__ void __launch_bounds__(256)
attn_kernel(const float* __restrict__ q, const float* __restrict__ k,
            const float* __restrict__ v, const float* __restrict__ ef,
            const int* __restrict__ src_idx, const int* __restrict__ dst_idx,
            float* __restrict__ out) {
  constexpr int C = HEADS * OH;
  static_assert(!STAGE || (NODES * C <= 65535), "tile_dim0 is 16-bit");
  extern __shared__ float sm[];
  float*        s_acc   = sm;                                        // NODES*C
  float*        s_logit = s_acc + NODES * C;                         // EDGES*HEADS
  unsigned int* s_max   = (unsigned int*)(s_logit + EDGES * HEADS);  // NODES*HEADS
  float*        s_sum   = (float*)(s_max + NODES * HEADS);           // NODES*HEADS
  float*        s_q     = s_sum + NODES * HEADS;                     // NODES*C (STAGE)
  float*        s_k     = s_q + NODES * C;
  float*        s_v     = s_k + NODES * C;

  const int g = blockIdx.x;
  const int tid = threadIdx.x;
  const int nodeBase = g * NODES;
  const int edgeBase = g * EDGES;

  if constexpr (STAGE) {
    if (tid < 32) {   // wave 0 issues the tensor DMAs and waits on its TENSORcnt
      tdm_load_1d(q + (size_t)nodeBase * C, s_q, NODES * C);
      tdm_load_1d(k + (size_t)nodeBase * C, s_k, NODES * C);
      tdm_load_1d(v + (size_t)nodeBase * C, s_v, NODES * C);
      __builtin_amdgcn_s_wait_tensorcnt(0);
    }
  }
  for (int i = tid; i < NODES * C; i += 256) s_acc[i] = 0.f;
  for (int i = tid; i < NODES * HEADS; i += 256) { s_max[i] = 0u; s_sum[i] = 0.f; }
  __syncthreads();

  const float scale = rsqrtf((float)OH);
  // pass 1: logits + per-(dst,head) max
  for (int i = tid; i < EDGES * HEADS; i += 256) {
    int e = i / HEADS, h = i % HEADS;
    int ge = edgeBase + e;
    int s = src_idx[ge], d = dst_idx[ge];
    const float* qd = STAGE ? (s_q + (size_t)(d - nodeBase) * C + h * OH)
                            : (q + (size_t)d * C + h * OH);
    const float* ks = STAGE ? (s_k + (size_t)(s - nodeBase) * C + h * OH)
                            : (k + (size_t)s * C + h * OH);
    const float* ep = HAS_EF ? (ef + (size_t)ge * C + h * OH) : nullptr;
    float dot = 0.f;
    for (int c = 0; c < OH; c += 4) {
      v4f qv = *(const v4f*)(qd + c);
      v4f kv = *(const v4f*)(ks + c);
      if (HAS_EF) {
        v4f ev = *(const v4f*)(ep + c);
        kv += ev;
      }
      dot += qv[0] * kv[0] + qv[1] * kv[1] + qv[2] * kv[2] + qv[3] * kv[3];
    }
    dot *= scale;
    float lg = (dot > 0.f) ? dot : 0.2f * dot;   // leaky_relu(., 0.2)
    s_logit[i] = lg;
    atomicMax(&s_max[(d - nodeBase) * HEADS + h], f2ord(lg));
  }
  __syncthreads();
  // pass 2: exp(logit - max), per-(dst,head) sum (ds_add_f32)
  for (int i = tid; i < EDGES * HEADS; i += 256) {
    int e = i / HEADS, h = i % HEADS;
    int d = dst_idx[edgeBase + e] - nodeBase;
    float m = ord2f(s_max[d * HEADS + h]);
    float ev = __expf(s_logit[i] - m);
    s_logit[i] = ev;
    atomicAdd(&s_sum[d * HEADS + h], ev);
  }
  __syncthreads();
  // pass 3: out[dst] += alpha * (v[src] (+ ef))   (scatter in LDS)
  for (int i = tid; i < EDGES * C; i += 256) {
    int e = i / C, c = i % C;
    int h = c / OH;
    int ge = edgeBase + e;
    int s = src_idx[ge];
    int d = dst_idx[ge] - nodeBase;
    float alpha = s_logit[e * HEADS + h] / (s_sum[d * HEADS + h] + 1e-16f);
    float vs = STAGE ? s_v[(size_t)(s - nodeBase) * C + c]
                     : v[(size_t)s * C + c];
    if (HAS_EF) vs += ef[(size_t)ge * C + c];
    atomicAdd(&s_acc[d * C + c], alpha * vs);
  }
  __syncthreads();
  for (int i = tid; i < NODES * C; i += 256)
    out[(size_t)nodeBase * C + i] = s_acc[i];
}

// ---------------- BatchNorm: column sums / apply -----------------------------
__global__ void __launch_bounds__(256)
colsum_kernel(const float* __restrict__ X, float* __restrict__ stats,
              int M, int N, int rowsPerBlock) {
  int col  = threadIdx.x % N;
  int sub  = threadIdx.x / N;
  int nsub = 256 / N;
  int r0 = blockIdx.x * rowsPerBlock;
  int r1 = r0 + rowsPerBlock; if (r1 > M) r1 = M;
  float s = 0.f, s2 = 0.f;
  for (int r = r0 + sub; r < r1; r += nsub) {
    float x = X[(size_t)r * N + col];
    s += x; s2 += x * x;
  }
  atomicAdd(&stats[col], s);
  atomicAdd(&stats[N + col], s2);
}

// out = subMode ? res - relu(bn(o)) : relu(bn(o)) + res
__global__ void __launch_bounds__(256)
bn_apply_kernel(const float* __restrict__ o, const float* __restrict__ res,
                const float* __restrict__ gamma, const float* __restrict__ beta,
                const float* __restrict__ stats, float* __restrict__ out,
                int M, int N, int subMode) {
  size_t idx = (size_t)blockIdx.x * 256 + threadIdx.x;
  size_t total = (size_t)M * N;
  if (idx >= total) return;
  int col = (int)(idx % N);
  float invM = 1.0f / (float)M;
  float mu  = stats[col] * invM;
  float var = stats[N + col] * invM - mu * mu;
  float val = (o[idx] - mu) * rsqrtf(var + 1e-5f) * gamma[col] + beta[col];
  val = fmaxf(val, 0.f);
  out[idx] = subMode ? (res[idx] - val) : (val + res[idx]);
}

// ---------------- BatchReversalToEdge (float4-wide) -------------------------
__global__ void __launch_bounds__(256)
to_edge_kernel(const float* __restrict__ br, float* __restrict__ out) {
  constexpr int V = CE / 4;                       // 16 float4 per row
  size_t idx = (size_t)blockIdx.x * 256 + threadIdx.x;
  if (idx >= (size_t)NEI * V) return;
  int c4  = (int)(idx % V);
  int row = (int)(idx / V);
  int g = row / EPG, r = row % EPG;
  v4f val;
  if (r < 2 * CNUM) {
    val = ((const v4f*)br)[((size_t)g * MPG + (r >> 1)) * V + c4];
  } else if (r < 2 * CNUM + NPG) {
    val = (v4f){0.f, 0.f, 0.f, 0.f};
    if (c4 == 0) val[0] = 1.0f;
  } else {
    val = ((const v4f*)br)[((size_t)g * MPG + (r - CNUM - NPG)) * V + c4];
  }
  ((v4f*)out)[idx] = val;
}

// ---------------- super node gather -----------------------------------------
__global__ void __launch_bounds__(256)
super_kernel(const float* __restrict__ out2, float* __restrict__ sn) {
  int idx = blockIdx.x * 256 + threadIdx.x;
  if (idx >= B * CN) return;
  int g = idx / CN, c = idx % CN;
  sn[idx] = out2[((size_t)g * NPG + (NPG - 1)) * CN + c];
}

// ---------------- host driver ------------------------------------------------
template <int K, int N>
static inline void gemm(const float* X, const float* W, float* Y, int M,
                        hipStream_t s) {
  dim3 grid(N / 64, M / 64);
  wmma_gemm_kernel<K, N><<<grid, 256, 0, s>>>(X, W, Y, M);
}

extern "C" void kernel_launch(void* const* d_in, const int* in_sizes, int n_in,
                              void* d_out, int out_size, void* d_ws, size_t ws_size,
                              hipStream_t stream) {
  (void)in_sizes; (void)n_in; (void)out_size; (void)ws_size;

  const float* x        = (const float*)d_in[0];
  const int*   ei_src   = (const int*)d_in[1];
  const int*   ei_dst   = ei_src + NEI;
  const float* br_in    = (const float*)d_in[2];
  const int*   be_src   = (const int*)d_in[3];
  const int*   be_dst   = be_src + NBE;
  const float* WqE1=(const float*)d_in[4],  *WkE1=(const float*)d_in[5],  *WvE1=(const float*)d_in[6];
  const float* gE1 =(const float*)d_in[7],  *bE1 =(const float*)d_in[8];
  const float* Wq1 =(const float*)d_in[9],  *Wk1 =(const float*)d_in[10], *Wv1 =(const float*)d_in[11];
  const float* We1 =(const float*)d_in[12], *gN1 =(const float*)d_in[13], *bN1 =(const float*)d_in[14];
  const float* WqE2=(const float*)d_in[15], *WkE2=(const float*)d_in[16], *WvE2=(const float*)d_in[17];
  const float* gE2 =(const float*)d_in[18], *bE2 =(const float*)d_in[19];
  const float* Wq2 =(const float*)d_in[20], *Wk2 =(const float*)d_in[21], *Wv2 =(const float*)d_in[22];
  const float* We2 =(const float*)d_in[23], *gN2 =(const float*)d_in[24], *bN2 =(const float*)d_in[25];

  // d_out slices: out2 | ei | br | super_node
  float* out2  = (float*)d_out;
  float* eiOut = out2 + (size_t)NN * CN;
  float* brOut = eiOut + (size_t)NEI * CE;
  float* snOut = brOut + (size_t)NBR * CE;

  // workspace carving
  float* ws = (float*)d_ws;
  size_t off = 0;
  float* brCur = ws + off; off += (size_t)NBR * CE;
  float* qE    = ws + off; off += (size_t)NBR * CE;
  float* kE    = ws + off; off += (size_t)NBR * CE;
  float* vE    = ws + off; off += (size_t)NBR * CE;
  float* oE    = ws + off; off += (size_t)NBR * CE;
  float* ei1   = ws + off; off += (size_t)NEI * CE;
  float* qN    = ws + off; off += (size_t)NN * CN;
  float* kN    = ws + off; off += (size_t)NN * CN;
  float* vN    = ws + off; off += (size_t)NN * CN;
  float* oN    = ws + off; off += (size_t)NN * CN;
  float* efB   = ws + off; off += (size_t)NEI * CN;
  float* out1  = ws + off; off += (size_t)NN * CN;
  float* stats = ws + off; off += 512;

  // dynamic LDS sizes
  const size_t smemE = ((size_t)MPG * CE + (size_t)(MPG * 4) * HE + 2u * MPG * HE
                        + 3u * MPG * CE) * 4;                    // ~140 KB (TDM-staged)
  const size_t smemN = ((size_t)NPG * CN + (size_t)EPG * HN + 2u * NPG * HN) * 4;

  // ---------------- layer E1 ----------------
  gemm<CE, CE>(br_in, WqE1, qE, NBR, stream);
  gemm<CE, CE>(br_in, WkE1, kE, NBR, stream);
  gemm<CE, CE>(br_in, WvE1, vE, NBR, stream);
  attn_kernel<MPG, MPG * 4, HE, CE / HE, false, true>
      <<<B, 256, smemE, stream>>>(qE, kE, vE, nullptr, be_src, be_dst, oE);
  hipMemsetAsync(stats, 0, 2 * CE * sizeof(float), stream);
  colsum_kernel<<<NBR / 256, 256, 0, stream>>>(oE, stats, NBR, CE, 256);
  bn_apply_kernel<<<(unsigned)(((size_t)NBR * CE + 255) / 256), 256, 0, stream>>>(
      oE, br_in, gE1, bE1, stats, brCur, NBR, CE, 0);
  to_edge_kernel<<<(unsigned)(((size_t)NEI * (CE / 4) + 255) / 256), 256, 0, stream>>>(brCur, ei1);

  // ---------------- layer N1 ----------------
  gemm<CN, CN>(x, Wq1, qN, NN, stream);
  gemm<CN, CN>(x, Wk1, kN, NN, stream);
  gemm<CN, CN>(x, Wv1, vN, NN, stream);
  gemm<CE, CN>(ei1, We1, efB, NEI, stream);
  attn_kernel<NPG, EPG, HN, CN / HN, true, false>
      <<<B, 256, smemN, stream>>>(qN, kN, vN, efB, ei_src, ei_dst, oN);
  hipMemsetAsync(stats, 0, 2 * CN * sizeof(float), stream);
  colsum_kernel<<<NN / 256, 256, 0, stream>>>(oN, stats, NN, CN, 256);
  bn_apply_kernel<<<(unsigned)(((size_t)NN * CN + 255) / 256), 256, 0, stream>>>(
      oN, x, gN1, bN1, stats, out1, NN, CN, 0);

  // ---------------- layer E2 (subtractive residual) ----------------
  gemm<CE, CE>(brCur, WqE2, qE, NBR, stream);
  gemm<CE, CE>(brCur, WkE2, kE, NBR, stream);
  gemm<CE, CE>(brCur, WvE2, vE, NBR, stream);
  attn_kernel<MPG, MPG * 4, HE, CE / HE, false, true>
      <<<B, 256, smemE, stream>>>(qE, kE, vE, nullptr, be_src, be_dst, oE);
  hipMemsetAsync(stats, 0, 2 * CE * sizeof(float), stream);
  colsum_kernel<<<NBR / 256, 256, 0, stream>>>(oE, stats, NBR, CE, 256);
  bn_apply_kernel<<<(unsigned)(((size_t)NBR * CE + 255) / 256), 256, 0, stream>>>(
      oE, brCur, gE2, bE2, stats, brOut, NBR, CE, 1);
  to_edge_kernel<<<(unsigned)(((size_t)NEI * (CE / 4) + 255) / 256), 256, 0, stream>>>(brOut, eiOut);

  // ---------------- layer N2 (subtractive residual) ----------------
  gemm<CN, CN>(out1, Wq2, qN, NN, stream);
  gemm<CN, CN>(out1, Wk2, kN, NN, stream);
  gemm<CN, CN>(out1, Wv2, vN, NN, stream);
  gemm<CE, CN>(eiOut, We2, efB, NEI, stream);
  attn_kernel<NPG, EPG, HN, CN / HN, true, false>
      <<<B, 256, smemN, stream>>>(qN, kN, vN, efB, ei_src, ei_dst, oN);
  hipMemsetAsync(stats, 0, 2 * CN * sizeof(float), stream);
  colsum_kernel<<<NN / 256, 256, 0, stream>>>(oN, stats, NN, CN, 256);
  bn_apply_kernel<<<(unsigned)(((size_t)NN * CN + 255) / 256), 256, 0, stream>>>(
      oN, out1, gN2, bN2, stats, out2, NN, CN, 1);

  // ---------------- super node ----------------
  super_kernel<<<(B * CN + 255) / 256, 256, 0, stream>>>(out2, snOut);
}
